// Attention_67233418051752
// MI455X (gfx1250) — matrix operand order repaired and checked by reading, not compiled, not run on hardware
//
#include <hip/hip_runtime.h>
#include <hip/hip_bf16.h>

// ---------------------------------------------------------------------------
// Types for WMMA fragments (wave32, CDNA5)
// ---------------------------------------------------------------------------
typedef _Float16 h8_t  __attribute__((ext_vector_type(8)));
typedef _Float16 h16_t __attribute__((ext_vector_type(16)));
typedef float    f8_t  __attribute__((ext_vector_type(8)));

__device__ __forceinline__ h16_t cat16(h8_t lo, h8_t hi) {
  return __builtin_shufflevector(lo, hi, 0,1,2,3,4,5,6,7,8,9,10,11,12,13,14,15);
}

// ---------------------------------------------------------------------------
// Problem constants
// ---------------------------------------------------------------------------
#define BSZ 4
#define SEQ 2048
#define CH  1024
#define NH  16
#define HD  64
#define MROWS (BSZ * SEQ)        // 8192
#define NEGBIG (-1.0e9f)

// ---------------------------------------------------------------------------
// f32 -> f16 conversion
// ---------------------------------------------------------------------------
__global__ __launch_bounds__(256) void cvt_f32_f16(const float* __restrict__ in,
                                                   _Float16* __restrict__ out, int n) {
  int i = blockIdx.x * blockDim.x + threadIdx.x;
  if (i < n) out[i] = (_Float16)in[i];
}

// ---------------------------------------------------------------------------
// GEMM: C[M,N] = A[M,K] * Bw[N,K]^T + bias[N]
// A, Bw are f16 row-major. One wave computes a 16x64 tile (A-frag reuse x4).
// A-frag (16-bit A 16x32): lanes 0-15 row=lr, K 0..15; lanes 16-31 K 16..31
// B-frag (16-bit B 32x16): lane col=lr, lanes 0-15 K 0..15, lanes 16-31 K 16..31;
//   column n of B == row n of Bw => contiguous 32B loads.
// D layout: n = lr, m = vgpr + 8*(lane>=16)
// ---------------------------------------------------------------------------
template <bool F16OUT>
__global__ __launch_bounds__(256) void gemm_wmma_kernel(
    const _Float16* __restrict__ A, const _Float16* __restrict__ Bw,
    const float* __restrict__ bias, void* __restrict__ Cout,
    int M, int N, int K)
{
  const int wave = (blockIdx.x * blockDim.x + threadIdx.x) >> 5;
  const int lane = threadIdx.x & 31;
  const int hf = lane >> 4, lr = lane & 15;
  const int ntiles = N >> 6;
  const int m0 = (wave / ntiles) << 4;
  const int n0 = (wave % ntiles) << 6;
  if (m0 >= M) return;

  f8_t acc[4] = {};
  const _Float16* arow = A + (size_t)(m0 + lr) * K;
  const _Float16* brow[4];
#pragma unroll
  for (int t = 0; t < 4; ++t) brow[t] = Bw + (size_t)(n0 + 16 * t + lr) * K;

  for (int k0 = 0; k0 < K; k0 += 32) {
    h16_t af = cat16(*(const h8_t*)(arow + k0 + 16 * hf),
                     *(const h8_t*)(arow + k0 + 8 + 16 * hf));
#pragma unroll
    for (int t = 0; t < 4; ++t) {
      h16_t bf = *(const h16_t*)(brow[t] + k0 + 16 * hf);
      acc[t] = __builtin_amdgcn_wmma_f32_16x16x32_f16(
          false, af, false, bf, (short)0, acc[t], false, false);
    }
  }

#pragma unroll
  for (int t = 0; t < 4; ++t) {
    const int col = n0 + 16 * t + lr;
    const float bv = bias[col];
#pragma unroll
    for (int r = 0; r < 8; ++r) {
      const int row = m0 + r + 8 * hf;
      const float v = acc[t][r] + bv;
      if (F16OUT)
        ((_Float16*)Cout)[(size_t)row * N + col] = (_Float16)v;
      else
        ((float*)Cout)[(size_t)row * N + col] = v;
    }
  }
}

// ---------------------------------------------------------------------------
// RoPE + reshape: Yqkv[8192][3072] (f16) ->
//   Q[bh][n][d] (rope, pre-scaled by 1/sqrt(D)), K[bh][n][d] (rope),
//   Vt[bh][d][n] (transposed).
// One thread per (b, n, h, d-pair).
// ---------------------------------------------------------------------------
__global__ __launch_bounds__(256) void rope_reorg_kernel(
    const _Float16* __restrict__ Yqkv,
    _Float16* __restrict__ Q, _Float16* __restrict__ Kq, _Float16* __restrict__ Vt)
{
  int idx = blockIdx.x * blockDim.x + threadIdx.x;       // (ng*16 + h)*32 + dp
  if (idx >= MROWS * NH * (HD / 2)) return;
  const int dp = idx & 31;
  const int h  = (idx >> 5) & 15;
  const int ng = idx >> 9;                               // global row 0..8191
  const int n  = ng & (SEQ - 1);
  const int b  = ng >> 11;

  const _Float16* src = Yqkv + (size_t)ng * (3 * CH) + h * HD + 2 * dp;
  const float q0 = (float)src[0],      q1 = (float)src[1];
  const float k0 = (float)src[CH],     k1 = (float)src[CH + 1];
  const float v0 = (float)src[2 * CH], v1 = (float)src[2 * CH + 1];

  // inv_freq = 10000^(-(2*dp)/64)
  const float inv = __expf(-((float)(2 * dp) / 64.0f) * 9.210340371976184f);
  const float ang = (float)n * inv;
  const float c = __cosf(ang), s = __sinf(ang);
  const float qs = 0.125f;   // 1/sqrt(64), folded into Q (exact power of 2)

  const size_t bh = (size_t)(b * NH + h);
  _Float16* qd = Q  + (bh * SEQ + n) * HD + 2 * dp;
  _Float16* kd = Kq + (bh * SEQ + n) * HD + 2 * dp;
  qd[0] = (_Float16)((q0 * c - q1 * s) * qs);
  qd[1] = (_Float16)((q0 * s + q1 * c) * qs);
  kd[0] = (_Float16)(k0 * c - k1 * s);
  kd[1] = (_Float16)(k0 * s + k1 * c);
  Vt[(bh * HD + 2 * dp)     * SEQ + n] = (_Float16)v0;
  Vt[(bh * HD + 2 * dp + 1) * SEQ + n] = (_Float16)v1;
}

// ---------------------------------------------------------------------------
// Flash attention, causal. One wave (32 threads) per (b, h, 16-query tile).
// Key blocks of 32. Per block: 4 WMMA for S = Q K^T, online softmax,
// P transposed through LDS into A-frag layout, 4 WMMA for P V.
// Causal mask applied only on the (rare) diagonal block.
// ---------------------------------------------------------------------------
__global__ __launch_bounds__(32) void flash_attn_kernel(
    const _Float16* __restrict__ Q, const _Float16* __restrict__ Kt,
    const _Float16* __restrict__ Vt, _Float16* __restrict__ Oh)
{
  __shared__ __align__(32) _Float16 lds[16 * 32];
  const int lane = threadIdx.x & 31;
  const int hf = lane >> 4, lr = lane & 15;
  const int tile = blockIdx.x;                  // 0 .. B*H*(N/16)-1
  const int qt = tile & 127;
  const int bh = tile >> 7;
  const int h  = bh & 15;
  const int b  = bh >> 4;
  const int q0 = qt << 4;

  const _Float16* Qb = Q  + (size_t)bh * SEQ * HD;
  const _Float16* Kb = Kt + (size_t)bh * SEQ * HD;
  const _Float16* Vb = Vt + (size_t)bh * HD * SEQ;

  // Preload Q A-fragments for d-chunks [0,32) and [32,64)
  h16_t qa[2];
  {
    const _Float16* qrow = Qb + (size_t)(q0 + lr) * HD;
#pragma unroll
    for (int c = 0; c < 2; ++c)
      qa[c] = cat16(*(const h8_t*)(qrow + 32 * c + 16 * hf),
                    *(const h8_t*)(qrow + 32 * c + 8 + 16 * hf));
  }

  f8_t acc[4] = {};
  float mrow[8], lrow[8];
#pragma unroll
  for (int r = 0; r < 8; ++r) { mrow[r] = -3.0e38f; lrow[r] = 0.0f; }
  const int nblocks = (q0 >> 5) + 1;

  for (int kb = 0; kb < nblocks; ++kb) {
    const int key0 = kb << 5;

    // Prefetch next key block's K rows (global_prefetch_b8)
    if (kb + 1 < nblocks) {
      const _Float16* knext = Kb + (size_t)(key0 + 32 + lane) * HD;
      __builtin_prefetch(knext, 0, 3);
    }

    f8_t s[2] = {};
#pragma unroll
    for (int sub = 0; sub < 2; ++sub) {
      const _Float16* krow = Kb + (size_t)(key0 + 16 * sub + lr) * HD + 16 * hf;
      h16_t kf0 = *(const h16_t*)(krow);        // d 0..31 chunk
      h16_t kf1 = *(const h16_t*)(krow + 32);   // d 32..63 chunk
      s[sub] = __builtin_amdgcn_wmma_f32_16x16x32_f16(
          false, qa[0], false, kf0, (short)0, s[sub], false, false);
      s[sub] = __builtin_amdgcn_wmma_f32_16x16x32_f16(
          false, qa[1], false, kf1, (short)0, s[sub], false, false);
    }

    // Causal mask — only the diagonal block can contain masked elements
    // (element (m,key): query = q0+r+8*hf, key = key0+16*sub+lr)
    if (key0 + 31 > q0) {
#pragma unroll
      for (int sub = 0; sub < 2; ++sub)
#pragma unroll
        for (int r = 0; r < 8; ++r) {
          if (key0 + 16 * sub + lr > q0 + r + 8 * hf) s[sub][r] = NEGBIG;
        }
    }

    // row max across the 16 lanes holding each row
    float pm[8];
#pragma unroll
    for (int r = 0; r < 8; ++r) pm[r] = fmaxf(s[0][r], s[1][r]);
#pragma unroll
    for (int off = 1; off < 16; off <<= 1)
#pragma unroll
      for (int r = 0; r < 8; ++r)
        pm[r] = fmaxf(pm[r], __shfl_xor(pm[r], off, 32));

    // online softmax update
    float ps[8], alpha[8];
#pragma unroll
    for (int r = 0; r < 8; ++r) {
      const float mn = fmaxf(mrow[r], pm[r]);
      alpha[r] = __expf(mrow[r] - mn);
      mrow[r] = mn;
      const float e0 = __expf(s[0][r] - mn);
      const float e1 = __expf(s[1][r] - mn);
      s[0][r] = e0; s[1][r] = e1;
      ps[r] = e0 + e1;
    }
#pragma unroll
    for (int off = 1; off < 16; off <<= 1)
#pragma unroll
      for (int r = 0; r < 8; ++r)
        ps[r] += __shfl_xor(ps[r], off, 32);
#pragma unroll
    for (int r = 0; r < 8; ++r) lrow[r] = lrow[r] * alpha[r] + ps[r];
#pragma unroll
    for (int t = 0; t < 4; ++t)
#pragma unroll
      for (int r = 0; r < 8; ++r) acc[t][r] *= alpha[r];

    // Transpose P (D-layout f32) -> A-frag layout (f16) through LDS
    __syncthreads();  // WAR vs previous iteration's fragment loads
#pragma unroll
    for (int sub = 0; sub < 2; ++sub)
#pragma unroll
      for (int r = 0; r < 8; ++r)
        lds[(r + 8 * hf) * 32 + 16 * sub + lr] = (_Float16)s[sub][r];
    __syncthreads();  // RAW: make stores visible to fragment loads
    h16_t pf = cat16(*(const h8_t*)(&lds[lr * 32 + 16 * hf]),
                     *(const h8_t*)(&lds[lr * 32 + 8 + 16 * hf]));

    // O += P * V  (V transposed: Vt row d contiguous along key)
#pragma unroll
    for (int t = 0; t < 4; ++t) {
      const _Float16* vrow = Vb + (size_t)(16 * t + lr) * SEQ + key0 + 16 * hf;
      h16_t vf = *(const h16_t*)vrow;
      acc[t] = __builtin_amdgcn_wmma_f32_16x16x32_f16(
          false, pf, false, vf, (short)0, acc[t], false, false);
    }
  }

  // Normalize and write Oh[b][n][h*64+d] (f16)
  _Float16* ob = Oh + ((size_t)(b * SEQ + q0)) * CH + h * HD;
#pragma unroll
  for (int r = 0; r < 8; ++r) {
    const float inv = 1.0f / lrow[r];
    const int m = r + 8 * hf;
#pragma unroll
    for (int t = 0; t < 4; ++t)
      ob[(size_t)m * CH + 16 * t + lr] = (_Float16)(acc[t][r] * inv);
  }
}

// ---------------------------------------------------------------------------
// Launcher
// ---------------------------------------------------------------------------
extern "C" void kernel_launch(void* const* d_in, const int* in_sizes, int n_in,
                              void* d_out, int out_size, void* d_ws, size_t ws_size,
                              hipStream_t stream) {
  const float* x       = (const float*)d_in[0];   // [4,2048,1024]
  const float* Wqkv_w  = (const float*)d_in[1];   // [3072,1024]
  const float* Wqkv_b  = (const float*)d_in[2];   // [3072]
  const float* out_w   = (const float*)d_in[3];   // [1024,1024]
  const float* out_b   = (const float*)d_in[4];   // [1024]
  float* out           = (float*)d_out;           // [4,2048,1024]

  char* ws = (char*)d_ws;
  // Workspace layout (bytes)
  _Float16* Xh   = (_Float16*)(ws + 0);                      // 16 MB  8192x1024
  _Float16* Wqh  = (_Float16*)(ws + (size_t)16777216);       // 6 MB   3072x1024
  _Float16* Woh  = (_Float16*)(ws + (size_t)23068672);       // 2 MB   1024x1024
  _Float16* Yqkv = (_Float16*)(ws + (size_t)25165824);       // 48 MB  8192x3072
  _Float16* Qb   = (_Float16*)(ws + (size_t)75497472);       // 16 MB  [64][2048][64]
  _Float16* Kb   = (_Float16*)(ws + (size_t)92274688);       // 16 MB
  _Float16* Vtb  = (_Float16*)(ws + (size_t)109051904);      // 16 MB  [64][64][2048]
  _Float16* Oh   = Yqkv;  // alias: Yqkv is dead after rope_reorg (8192x1024 f16)

  // 1) f32 -> f16 conversions
  {
    int n = MROWS * CH;            // 8388608
    cvt_f32_f16<<<(n + 255) / 256, 256, 0, stream>>>(x, Xh, n);
    n = 3 * CH * CH;               // 3145728
    cvt_f32_f16<<<(n + 255) / 256, 256, 0, stream>>>(Wqkv_w, Wqh, n);
    n = CH * CH;                   // 1048576
    cvt_f32_f16<<<(n + 255) / 256, 256, 0, stream>>>(out_w, Woh, n);
  }

  // 2) QKV projection: Yqkv = Xh * Wqh^T + b   (8192 x 3072)
  {
    int waves = (MROWS / 16) * (3 * CH / 64);    // 24576
    gemm_wmma_kernel<true><<<waves * 32 / 256, 256, 0, stream>>>(
        Xh, Wqh, Wqkv_b, (void*)Yqkv, MROWS, 3 * CH, CH);
  }

  // 3) RoPE + reshape to Q/K/Vt
  {
    int n = MROWS * NH * (HD / 2);               // 4194304
    rope_reorg_kernel<<<(n + 255) / 256, 256, 0, stream>>>(Yqkv, Qb, Kb, Vtb);
  }

  // 4) Flash attention -> Oh (f16, [8192][1024])
  {
    int blocks = BSZ * NH * (SEQ / 16);          // 8192
    flash_attn_kernel<<<blocks, 32, 0, stream>>>(Qb, Kb, Vtb, Oh);
  }

  // 5) Output projection: out = Oh * Woh^T + out_b   (8192 x 1024, f32)
  {
    int waves = (MROWS / 16) * (CH / 64);        // 8192
    gemm_wmma_kernel<false><<<waves * 32 / 256, 256, 0, stream>>>(
        Oh, Woh, out_b, (void*)out, MROWS, CH, CH);
  }
}
